// FastTransformer_28028956573924
// MI455X (gfx1250) — matrix operand order repaired and checked by reading, not compiled, not run on hardware
//
#include <hip/hip_runtime.h>
#include <hip/hip_bf16.h>
#include <cmath>

// ---------------------------------------------------------------------------
// Types for CDNA5 WMMA
// ---------------------------------------------------------------------------
typedef _Float16 half_t;
typedef __attribute__((ext_vector_type(16))) _Float16 v16h;
typedef __attribute__((ext_vector_type(8)))  _Float16 v8h;
typedef __attribute__((ext_vector_type(8)))  float    v8f;

#define DIMD   512
#define NSEQ   4096
#define HEADS  8
#define DH     64
#define INNER  512
#define FFD    2048
#define VOCAB  32000
#define SCALE_ATTN 0.125f   // DH^-0.5
#define EPS_LN 0.001f

#if __has_builtin(__builtin_amdgcn_global_load_async_to_lds_b128) && \
    __has_builtin(__builtin_amdgcn_s_wait_asynccnt)
#define USE_ASYNC_LDS 1
// Builtin signature (from hipcc diagnostics): params are pointers to a 128-bit
// int vector, global (AS1) source and LDS (AS3) destination, both non-const.
typedef int v4i_vec __attribute__((vector_size(16)));
typedef v4i_vec __attribute__((address_space(1)))* gasync_p;
typedef v4i_vec __attribute__((address_space(3)))* lasync_p;
#else
#define USE_ASYNC_LDS 0
#endif

// ---------------------------------------------------------------------------
// Weight convert + K-tile pack:  in[K,N] (f32, row-major)
//   -> out[(k/32)*N*32 + n*32 + (k%32)] (f16)
// Makes every 128x32 B tile a single contiguous 8KB chunk for the GEMM.
// ---------------------------------------------------------------------------
__global__ __launch_bounds__(256) void pack_b_kernel(const float* __restrict__ in,
                                                     half_t* __restrict__ out,
                                                     int K, int N) {
  int idx = blockIdx.x * 256 + threadIdx.x;
  if (idx >= K * N) return;
  int kk  = idx & 31;
  int rem = idx >> 5;
  int n   = rem % N;
  int kt  = rem / N;
  out[idx] = (half_t)in[(size_t)(kt * 32 + kk) * N + n];
}

// ---------------------------------------------------------------------------
// Embedding gather: x[m, :] = token_emb[tokens[m], :]
// ---------------------------------------------------------------------------
__global__ __launch_bounds__(256) void embed_kernel(const int* __restrict__ tokens,
                                                    const float* __restrict__ emb,
                                                    float* __restrict__ x) {
  int m = blockIdx.x;
  int t = tokens[m];
  int d = threadIdx.x;
  x[(size_t)m * DIMD + d]       = emb[(size_t)t * DIMD + d];
  x[(size_t)m * DIMD + d + 256] = emb[(size_t)t * DIMD + d + 256];
}

// ---------------------------------------------------------------------------
// LayerNorm over D=512, one block per row, writes f16 (feeds WMMA A operand)
// ---------------------------------------------------------------------------
__global__ __launch_bounds__(256) void ln_kernel(const float* __restrict__ x,
                                                 const float* __restrict__ scale,
                                                 const float* __restrict__ bias,
                                                 half_t* __restrict__ out) {
  __shared__ float red[256];
  const int row = blockIdx.x;
  const int tid = threadIdx.x;
  const float* xr = x + (size_t)row * DIMD;
  float v0 = xr[tid], v1 = xr[tid + 256];

  red[tid] = v0 + v1;
  __syncthreads();
  for (int o = 128; o > 0; o >>= 1) { if (tid < o) red[tid] += red[tid + o]; __syncthreads(); }
  float mu = red[0] * (1.0f / DIMD);
  __syncthreads();

  float d0 = v0 - mu, d1 = v1 - mu;
  red[tid] = d0 * d0 + d1 * d1;
  __syncthreads();
  for (int o = 128; o > 0; o >>= 1) { if (tid < o) red[tid] += red[tid + o]; __syncthreads(); }
  float rs = __frsqrt_rn(red[0] * (1.0f / DIMD) + EPS_LN);

  out[(size_t)row * DIMD + tid]       = (half_t)(d0 * rs * scale[tid]       + bias[tid]);
  out[(size_t)row * DIMD + tid + 256] = (half_t)(d1 * rs * scale[tid + 256] + bias[tid + 256]);
}

// ---------------------------------------------------------------------------
// WMMA GEMM:  C[M,N] = A[M,K](f16) * Bpacked  (+bias +resid, gelu, f32/f16 out)
// 128x128x32 block tile, 256 threads = 8 waves, wave tile 32x64 = 2x4 WMMAs.
// Double-buffered LDS; async global->LDS deposits when the toolchain has them.
// ---------------------------------------------------------------------------
#define BM 128
#define BN 128
#define BK 32
#define LDT 40   // padded LDS stride in halfs (lane bank spacing of 4 -> conflict-free)

__global__ __launch_bounds__(256) void gemm_wmma_kernel(
    const half_t* __restrict__ A, const half_t* __restrict__ B,
    const float* __restrict__ bias, const float* __restrict__ resid,
    float* __restrict__ outF, half_t* __restrict__ outH,
    int M, int N, int K, int gelu) {
  __shared__ __align__(16) half_t As[2][BM * LDT];
  __shared__ __align__(16) half_t Bs[2][BN * LDT];

  const int tid  = threadIdx.x;
  const int lane = tid & 31;
  const int wave = tid >> 5;
  const int wm   = wave >> 1;          // 0..3  -> 32-row strip
  const int wn   = wave & 1;           // 0..1  -> 64-col strip
  const int bn   = blockIdx.x * BN;
  const int bm   = blockIdx.y * BM;

  v8f acc[2][4] = {};

  // staging slots: each thread owns 4 x 16B chunks (2 for A, 2 for B)
  const int ld_row = tid >> 2;               // 0..63
  const int ld_off = (tid & 3) * 8;
  const half_t* gA0 = A + (size_t)(bm + ld_row) * K + ld_off;
  const half_t* gA1 = gA0 + (size_t)64 * K;
  const half_t* gB0 = B + ((size_t)bn + ld_row) * 32 + ld_off;
  const half_t* gB1 = gB0 + 64 * 32;
  const size_t  bStep = (size_t)N * 32;      // packed-B halfs-per-ktile
  const int     lA0 = ld_row * LDT + ld_off, lA1 = (ld_row + 64) * LDT + ld_off;

  auto compute = [&](int b) {
    const int ko = (lane >> 4) * 16;   // A: elems 0-15 = K[ko..ko+15]
    const int ar = lane & 15;
    v16h af[2];
#pragma unroll
    for (int mi = 0; mi < 2; ++mi) {
      int row = wm * 32 + mi * 16 + ar;
      v8h lo = *(const v8h*)(&As[b][row * LDT + ko]);
      v8h hi = *(const v8h*)(&As[b][row * LDT + ko + 8]);
      af[mi] = __builtin_shufflevector(lo, hi, 0,1,2,3,4,5,6,7,8,9,10,11,12,13,14,15);
    }
    const int kb2 = (lane >> 4) * 16;  // B: elems 0-15 = K[kb2..kb2+15]
    const int bc  = lane & 15;
#pragma unroll
    for (int ni = 0; ni < 4; ++ni) {
      int col = wn * 64 + ni * 16 + bc;
      v8h lo = *(const v8h*)(&Bs[b][col * LDT + kb2]);
      v8h hi = *(const v8h*)(&Bs[b][col * LDT + kb2 + 8]);
      v16h bf = __builtin_shufflevector(lo, hi, 0,1,2,3,4,5,6,7,8,9,10,11,12,13,14,15);
#pragma unroll
      for (int mi = 0; mi < 2; ++mi)
        acc[mi][ni] = __builtin_amdgcn_wmma_f32_16x16x32_f16(
            false, af[mi], false, bf, (short)0, acc[mi][ni], false, false);
    }
  };

#if USE_ASYNC_LDS
  auto stage = [&](int kt, int b) {     // 4 async-to-LDS deposits per thread
    const size_t bo = (size_t)(kt >> 5) * bStep;
    __builtin_amdgcn_global_load_async_to_lds_b128(
        (gasync_p)(gA0 + kt), (lasync_p)&As[b][lA0], 0, 0);
    __builtin_amdgcn_global_load_async_to_lds_b128(
        (gasync_p)(gA1 + kt), (lasync_p)&As[b][lA1], 0, 0);
    __builtin_amdgcn_global_load_async_to_lds_b128(
        (gasync_p)(gB0 + bo), (lasync_p)&Bs[b][lA0], 0, 0);
    __builtin_amdgcn_global_load_async_to_lds_b128(
        (gasync_p)(gB1 + bo), (lasync_p)&Bs[b][lA1], 0, 0);
  };

  stage(0, 0);
  int i = 0;
  for (int kt = 0; kt < K; kt += BK, ++i) {
    const bool more = (kt + BK) < K;
    if (more) stage(kt + BK, (i + 1) & 1);   // next tile in flight during compute
    if (more) __builtin_amdgcn_s_wait_asynccnt(4);   // newest 4 = next tile; current done
    else      __builtin_amdgcn_s_wait_asynccnt(0);
    __syncthreads();                         // all waves' deposits visible
    compute(i & 1);
    __syncthreads();                         // readers done before buffer is re-staged
  }
#else
  v8h ra0, ra1, rb0, rb1;
  auto load_regs = [&](int kt) {
    const size_t bo = (size_t)(kt >> 5) * bStep;
    ra0 = *(const v8h*)(gA0 + kt);
    ra1 = *(const v8h*)(gA1 + kt);
    rb0 = *(const v8h*)(gB0 + bo);
    rb1 = *(const v8h*)(gB1 + bo);
    if (kt + BK < K) {
      __builtin_prefetch((const void*)(gA0 + kt + BK), 0, 3);
      __builtin_prefetch((const void*)(gB0 + bo + bStep), 0, 3);
    }
  };
  auto store_regs = [&](int b) {
    *(v8h*)(&As[b][lA0]) = ra0;
    *(v8h*)(&As[b][lA1]) = ra1;
    *(v8h*)(&Bs[b][lA0]) = rb0;
    *(v8h*)(&Bs[b][lA1]) = rb1;
  };
  load_regs(0);
  store_regs(0);
  int i = 0;
  for (int kt = 0; kt < K; kt += BK, ++i) {
    __syncthreads();
    const bool more = (kt + BK) < K;
    if (more) load_regs(kt + BK);            // global latency overlapped with WMMAs
    compute(i & 1);
    if (more) store_regs((i + 1) & 1);       // waits loadcnt only here
  }
#endif

  // ---- epilogue ----
  const int lr = lane >> 4;  // M = v + 8*lr
  const int lc = lane & 15;  // N = lc
#pragma unroll
  for (int mi = 0; mi < 2; ++mi) {
#pragma unroll
    for (int ni = 0; ni < 4; ++ni) {
      int gcol = bn + wn * 64 + ni * 16 + lc;
      float bv = bias ? bias[gcol] : 0.0f;
#pragma unroll
      for (int v = 0; v < 8; ++v) {
        int grow = bm + wm * 32 + mi * 16 + lr * 8 + v;
        size_t off = (size_t)grow * N + gcol;
        float val = acc[mi][ni][v] + bv;
        if (resid) val += resid[off];
        if (gelu)  val = 0.5f * val * (1.0f + erff(val * 0.70710678f));
        if (outF)  outF[off] = val;
        if (outH)  outH[off] = (half_t)val;
      }
    }
  }
}

// ---------------------------------------------------------------------------
// Linear ("fast") attention core: one block per head.
// qkv layout: [n, 1536] with q = [:,h*64+d], k = +512, v = +1024.
// Writes r (f16) at out[n, head*64 + dh]  (feeds the w_out GEMM).
// ---------------------------------------------------------------------------
__device__ __forceinline__ float block_reduce(float v, float* red, int tid, bool ismax) {
  red[tid] = v;
  __syncthreads();
  for (int o = 128; o > 0; o >>= 1) {
    if (tid < o) red[tid] = ismax ? fmaxf(red[tid], red[tid + o]) : (red[tid] + red[tid + o]);
    __syncthreads();
  }
  float r = red[0];
  __syncthreads();
  return r;
}

__device__ __forceinline__ float rot_elem(const float* __restrict__ row, int n, int d,
                                          const float* __restrict__ invf) {
  if (d >= 32) return row[d];
  float th = (float)n * invf[d >> 1];
  float c = __cosf(th), s = __sinf(th);
  return (d & 1) ? (row[d] * c + row[d - 1] * s)
                 : (row[d] * c - row[d + 1] * s);
}

__global__ __launch_bounds__(256) void attn_kernel(
    const float* __restrict__ qkv,
    const float* __restrict__ wq_logit, const float* __restrict__ wk_logit,
    const float* __restrict__ w_r, const float* __restrict__ b_r,
    half_t* __restrict__ out) {
  __shared__ float s_s[NSEQ];        // scores / weights (16 KB)
  __shared__ float s_red[256];
  __shared__ float s_wq[64];
  __shared__ float s_wk[32];
  __shared__ float s_br[64];
  __shared__ float s_wr[32 * 64];    // 8 KB
  __shared__ float s_g[64];          // global_q then global_k
  __shared__ float s_gw[64];         // global_q * wk_logit[d/2]
  __shared__ float s_acc[4][64];
  __shared__ float s_invf[16];

  const int head = blockIdx.x;
  const int tid  = threadIdx.x;

  if (tid < 64) s_wq[tid] = wq_logit[tid];
  if (tid < 32) s_wk[tid] = wk_logit[tid];
  if (tid < 64) s_br[tid] = b_r[tid];
  if (tid < 16) s_invf[tid] = __expf(-(float)(2 * tid) * (9.2103403720f / 32.0f));
  for (int i = tid; i < 32 * 64; i += 256) s_wr[i] = w_r[i];
  __syncthreads();

  const float* qb = qkv + head * DH;
  const float* kb = qkv + 512 + head * DH;
  const float* vb = qkv + 1024 + head * DH;

  // ================= Phase Q: q_attn softmax & global_q =================
  {
    float lmax = -1e30f;
    for (int n = tid; n < NSEQ; n += 256) {
      const float* row = qb + (size_t)n * 1536;
      float acc = 0.f;
#pragma unroll
      for (int d = 0; d < 64; ++d) acc += row[d] * s_wq[d];
      acc *= SCALE_ATTN;
      s_s[n] = acc;
      lmax = fmaxf(lmax, acc);
    }
    float m = block_reduce(lmax, s_red, tid, true);
    float lsum = 0.f;
    for (int n = tid; n < NSEQ; n += 256) {
      float e = __expf(s_s[n] - m);
      s_s[n] = e;
      lsum += e;
    }
    float sum = block_reduce(lsum, s_red, tid, false);

    int d = tid & 63, g = tid >> 6;
    float acc = 0.f;
    for (int n = g; n < NSEQ; n += 4)
      acc += s_s[n] * rot_elem(qb + (size_t)n * 1536, n, d, s_invf);
    s_acc[g][d] = acc;
    __syncthreads();
    if (g == 0) {
      float gv = (s_acc[0][d] + s_acc[1][d] + s_acc[2][d] + s_acc[3][d]) / sum;
      s_g[d]  = gv;
      s_gw[d] = gv * s_wk[d >> 1];
    }
    __syncthreads();
  }

  // ================= Phase K: k_attn softmax & global_k =================
  {
    float lmax = -1e30f;
    for (int n = tid; n < NSEQ; n += 256) {
      const float* row = kb + (size_t)n * 1536;
      float acc = 0.f;
#pragma unroll
      for (int d = 0; d < 64; ++d) acc += row[d] * s_gw[d];
      acc *= SCALE_ATTN;
      s_s[n] = acc;
      lmax = fmaxf(lmax, acc);
    }
    float m = block_reduce(lmax, s_red, tid, true);
    float lsum = 0.f;
    for (int n = tid; n < NSEQ; n += 256) {
      float e = __expf(s_s[n] - m);
      s_s[n] = e;
      lsum += e;
    }
    float sum = block_reduce(lsum, s_red, tid, false);

    int d = tid & 63, g = tid >> 6;
    float acc = 0.f;
    for (int n = g; n < NSEQ; n += 4)
      acc += s_s[n] * rot_elem(kb + (size_t)n * 1536, n, d, s_invf);
    s_acc[g][d] = acc;
    __syncthreads();
    if (g == 0) s_g[d] = (s_acc[0][d] + s_acc[1][d] + s_acc[2][d] + s_acc[3][d]) / sum;
    __syncthreads();
  }

  // ========== Phase V/R: u = pairsum(v_rot * gk); r = u @ w_r + b_r + q ==========
  for (int n = tid; n < NSEQ; n += 256) {
    const float* vrow = vb + (size_t)n * 1536;
    const float* qrow = qb + (size_t)n * 1536;
    float u[32];
#pragma unroll
    for (int p = 0; p < 32; ++p) {
      float e0 = rot_elem(vrow, n, 2 * p, s_invf);
      float e1 = rot_elem(vrow, n, 2 * p + 1, s_invf);
      u[p] = e0 * s_g[2 * p] + e1 * s_g[2 * p + 1];
    }
    half_t* orow = out + (size_t)n * INNER + head * DH;
#pragma unroll 4
    for (int dh = 0; dh < 64; ++dh) {
      float acc = s_br[dh] + qrow[dh];
#pragma unroll
      for (int p = 0; p < 32; ++p) acc += u[p] * s_wr[p * 64 + dh];
      orow[dh] = (half_t)acc;
    }
  }
}

// ---------------------------------------------------------------------------
// Host-side orchestration
// ---------------------------------------------------------------------------
extern "C" void kernel_launch(void* const* d_in, const int* in_sizes, int n_in,
                              void* d_out, int out_size, void* d_ws, size_t ws_size,
                              hipStream_t stream) {
  const int*   tokens         = (const int*)d_in[0];
  const float* token_emb      = (const float*)d_in[1];
  const float* attn_ln_scale  = (const float*)d_in[2];
  const float* attn_ln_bias   = (const float*)d_in[3];
  const float* w_qkv          = (const float*)d_in[4];
  const float* wq_logit       = (const float*)d_in[5];
  const float* wk_logit       = (const float*)d_in[6];
  const float* w_r            = (const float*)d_in[7];
  const float* b_r            = (const float*)d_in[8];
  const float* w_out          = (const float*)d_in[9];
  const float* b_out          = (const float*)d_in[10];
  const float* ff_ln_scale    = (const float*)d_in[11];
  const float* ff_ln_bias     = (const float*)d_in[12];
  const float* w_ff1          = (const float*)d_in[13];
  const float* b_ff1          = (const float*)d_in[14];
  const float* w_ff2          = (const float*)d_in[15];
  const float* b_ff2          = (const float*)d_in[16];
  const float* final_ln_scale = (const float*)d_in[17];
  const float* final_ln_bias  = (const float*)d_in[18];
  const float* w_logits       = (const float*)d_in[19];
  const float* b_logits       = (const float*)d_in[20];

  // ---- workspace carve-out (256B aligned) ----
  char* wsp = (char*)d_ws;
  auto alloc = [&](size_t bytes) -> void* {
    void* p = (void*)wsp;
    wsp += (bytes + 255) & ~(size_t)255;
    return p;
  };
  float*  xA     = (float*)alloc((size_t)NSEQ * DIMD * 4);
  float*  xB     = (float*)alloc((size_t)NSEQ * DIMD * 4);
  half_t* hH     = (half_t*)alloc((size_t)NSEQ * DIMD * 2);
  float*  qkvF   = (float*)alloc((size_t)NSEQ * 3 * INNER * 4);
  half_t* attnH  = (half_t*)alloc((size_t)NSEQ * INNER * 2);
  half_t* ff1H   = (half_t*)alloc((size_t)NSEQ * FFD * 2);
  half_t* wqkvH  = (half_t*)alloc((size_t)2 * DIMD * 3 * INNER * 2);
  half_t* woutH  = (half_t*)alloc((size_t)2 * INNER * DIMD * 2);
  half_t* wff1H  = (half_t*)alloc((size_t)2 * DIMD * FFD * 2);
  half_t* wff2H  = (half_t*)alloc((size_t)2 * FFD * DIMD * 2);
  half_t* wlogH  = (half_t*)alloc((size_t)DIMD * VOCAB * 2);

  auto packB = [&](const float* src, half_t* dst, int K, int N) {
    size_t n = (size_t)K * N;
    pack_b_kernel<<<dim3((unsigned)((n + 255) / 256)), dim3(256), 0, stream>>>(src, dst, K, N);
  };
  packB(w_qkv,    wqkvH, 2 * DIMD, 3 * INNER);   // per-layer slices stay contiguous
  packB(w_out,    woutH, 2 * INNER, DIMD);
  packB(w_ff1,    wff1H, 2 * DIMD, FFD);
  packB(w_ff2,    wff2H, 2 * FFD, DIMD);
  packB(w_logits, wlogH, DIMD, VOCAB);

  embed_kernel<<<dim3(NSEQ), dim3(256), 0, stream>>>(tokens, token_emb, xA);

  for (int l = 0; l < 2; ++l) {
    // --- attention block ---
    ln_kernel<<<dim3(NSEQ), dim3(256), 0, stream>>>(xA, attn_ln_scale + l * DIMD,
                                                    attn_ln_bias + l * DIMD, hH);
    gemm_wmma_kernel<<<dim3(3 * INNER / BN, NSEQ / BM), dim3(256), 0, stream>>>(
        hH, wqkvH + (size_t)l * DIMD * 3 * INNER, nullptr, nullptr,
        qkvF, nullptr, NSEQ, 3 * INNER, DIMD, 0);
    attn_kernel<<<dim3(HEADS), dim3(256), 0, stream>>>(
        qkvF, wq_logit, wk_logit, w_r + (size_t)l * 32 * 64, b_r + (size_t)l * 64, attnH);
    gemm_wmma_kernel<<<dim3(DIMD / BN, NSEQ / BM), dim3(256), 0, stream>>>(
        attnH, woutH + (size_t)l * INNER * DIMD, b_out + l * DIMD, xA,
        xB, nullptr, NSEQ, DIMD, INNER, 0);

    // --- feed-forward block ---
    ln_kernel<<<dim3(NSEQ), dim3(256), 0, stream>>>(xB, ff_ln_scale + l * DIMD,
                                                    ff_ln_bias + l * DIMD, hH);
    gemm_wmma_kernel<<<dim3(FFD / BN, NSEQ / BM), dim3(256), 0, stream>>>(
        hH, wff1H + (size_t)l * DIMD * FFD, b_ff1 + l * FFD, nullptr,
        nullptr, ff1H, NSEQ, FFD, DIMD, 1 /*gelu*/);
    gemm_wmma_kernel<<<dim3(DIMD / BN, NSEQ / BM), dim3(256), 0, stream>>>(
        ff1H, wff2H + (size_t)l * FFD * DIMD, b_ff2 + l * DIMD, xB,
        xA, nullptr, NSEQ, DIMD, FFD, 0);
  }

  // --- final LN + vocab projection (the 134 GFLOP GEMM) ---
  ln_kernel<<<dim3(NSEQ), dim3(256), 0, stream>>>(xA, final_ln_scale, final_ln_bias, hH);
  gemm_wmma_kernel<<<dim3(VOCAB / BN, NSEQ / BM), dim3(256), 0, stream>>>(
      hH, wlogH, b_logits, nullptr, (float*)d_out, nullptr, NSEQ, VOCAB, DIMD, 0);
}
